// SwappedTransformerLayerRoPE_51883204936010
// MI455X (gfx1250) — matrix operand order repaired and checked by reading, not compiled, not run on hardware
//
#include <hip/hip_runtime.h>
#include <hip/hip_bf16.h>

typedef __attribute__((ext_vector_type(16))) _Float16 v16h;
typedef __attribute__((ext_vector_type(8)))  _Float16 v8h;
typedef __attribute__((ext_vector_type(8)))  float    v8f;

#define D_MODEL 1024
#define SEQ     512
#define BATCH   8
#define NHEAD   16
#define HDIM    64
#define MTOT    (BATCH * SEQ)   // 4096

union AFrag { v16h v; v8h h[2]; };

__device__ __forceinline__ v8f zerov8() {
    v8f z;
#pragma unroll
    for (int i = 0; i < 8; i++) z[i] = 0.0f;
    return z;
}

__device__ __forceinline__ v8f wmma_f16(const v16h& a, const v16h& b, const v8f& c) {
    // D = A(16x32 f16) * B(32x16 f16) + C(16x16 f32)
    return __builtin_amdgcn_wmma_f32_16x16x32_f16(false, a, false, b, (short)0, c, false, false);
}

// ---------------------------------------------------------------------------
// Weight convert: W[k,n] f32 row-major -> Wt[n,k] f16 row-major (transposed)
// ---------------------------------------------------------------------------
__global__ __launch_bounds__(256) void wconv_kernel(const float* __restrict__ W,
                                                    _Float16* __restrict__ Wt) {
    int idx = blockIdx.x * 256 + threadIdx.x;   // 0 .. 1024*1024-1
    int n = idx >> 10;
    int k = idx & 1023;
    Wt[idx] = (_Float16)W[((size_t)k << 10) + n];
}

// ---------------------------------------------------------------------------
// LayerNorm: per row of 1024, out f16
// ---------------------------------------------------------------------------
__global__ __launch_bounds__(256) void ln_kernel(const float* __restrict__ x,
                                                 const float* __restrict__ g,
                                                 const float* __restrict__ bb,
                                                 _Float16* __restrict__ xn) {
    __shared__ float red[16];
    int row = blockIdx.x;
    int t = threadIdx.x;
    const float* xr = x + (size_t)row * D_MODEL;
    float4 v = ((const float4*)xr)[t];
    float s  = v.x + v.y + v.z + v.w;
    float sq = v.x * v.x + v.y * v.y + v.z * v.z + v.w * v.w;
#pragma unroll
    for (int off = 16; off; off >>= 1) {
        s  += __shfl_down(s, off, 32);
        sq += __shfl_down(sq, off, 32);
    }
    int wave = t >> 5, lane = t & 31;
    if (lane == 0) { red[wave] = s; red[8 + wave] = sq; }
    __syncthreads();
    if (t == 0) {
        float S = 0.f, SQ = 0.f;
        for (int i = 0; i < 8; i++) { S += red[i]; SQ += red[8 + i]; }
        red[0] = S; red[8] = SQ;
    }
    __syncthreads();
    float mean = red[0] * (1.0f / D_MODEL);
    float var  = red[8] * (1.0f / D_MODEL) - mean * mean;
    float rs   = rsqrtf(var + 1e-5f);
    int c0 = t * 4;
    _Float16* o = xn + (size_t)row * D_MODEL + c0;
    const float* gp = g + c0;
    const float* bp = bb + c0;
    o[0] = (_Float16)((v.x - mean) * rs * gp[0] + bp[0]);
    o[1] = (_Float16)((v.y - mean) * rs * gp[1] + bp[1]);
    o[2] = (_Float16)((v.z - mean) * rs * gp[2] + bp[2]);
    o[3] = (_Float16)((v.w - mean) * rs * gp[3] + bp[3]);
}

// ---------------------------------------------------------------------------
// Generic WMMA GEMM: C[M=4096, N=1024] = A_f16[M,1024] @ (Bt_f16[N,1024])^T + bias
// One wave computes a 32x64 tile (2 M row-blocks reuse each B fragment).
// 8 waves/block -> 256 blocks. 8 WMMA per 32-K step.
// MODE 0: Q epilogue (RoPE + 0.125 scale, write f16 [bh][l][hd])
// MODE 1: K epilogue (RoPE,              write f16 [bh][l][hd])
// MODE 2: V epilogue (plain,             write f16 [bh][hd][l]   == V^T)
// MODE 3: O epilogue (bias + residual,   write f32 [M][N] to d_out)
// ---------------------------------------------------------------------------
template <int MODE>
__global__ __launch_bounds__(256) void gemm_kernel(const _Float16* __restrict__ A,
                                                   const _Float16* __restrict__ Bt,
                                                   const float* __restrict__ bias,
                                                   const float* __restrict__ resid,
                                                   void* __restrict__ outp) {
    const int lane = threadIdx.x & 31;
    const int wave = threadIdx.x >> 5;
    const int tile = blockIdx.x * 8 + wave;   // 0..2047
    const int mt = tile >> 4;                 // 0..127  (32-row block)
    const int ng = tile & 15;                 // 0..15   (64-col group)
    const int m0 = mt * 32;
    const int n0 = ng * 64;
    const int l15 = lane & 15;
    const int hiw = lane >> 4;                // 0 or 1

    v8f acc[2][4];
#pragma unroll
    for (int mm = 0; mm < 2; mm++)
#pragma unroll
        for (int t = 0; t < 4; t++) acc[mm][t] = zerov8();

    // A fragment base: row = m0 + mm*16 + l15; halves 0..15 = K(hiw*16..) contiguous
    const _Float16* Ab0 = A + (size_t)(m0 + l15) * 1024 + hiw * 16;
    const _Float16* Ab1 = Ab0 + (size_t)16 * 1024;
    // B fragment: per-lane 16 contiguous K values at (hiw*16)
    const _Float16* Bb = Bt + hiw * 16;

    for (int k0 = 0; k0 < 1024; k0 += 32) {
        // prefetch next K-step of the A stream (global_prefetch_b8)
        __builtin_prefetch((const void*)(Ab0 + k0 + 32), 0, 1);
        __builtin_prefetch((const void*)(Ab1 + k0 + 32), 0, 1);
        AFrag a0, a1;
        a0.h[0] = *(const v8h*)(Ab0 + k0);
        a0.h[1] = *(const v8h*)(Ab0 + k0 + 8);
        a1.h[0] = *(const v8h*)(Ab1 + k0);
        a1.h[1] = *(const v8h*)(Ab1 + k0 + 8);
#pragma unroll
        for (int t = 0; t < 4; t++) {
            int col = n0 + t * 16 + l15;
            v16h bf = *(const v16h*)(Bb + (size_t)col * 1024 + k0);
            acc[0][t] = wmma_f16(a0.v, bf, acc[0][t]);
            acc[1][t] = wmma_f16(a1.v, bf, acc[1][t]);
        }
    }

#pragma unroll
    for (int mm = 0; mm < 2; mm++) {
        const int mb = m0 + mm * 16;
        if (MODE == 3) {
            float* out = (float*)outp;
#pragma unroll
            for (int t = 0; t < 4; t++) {
                int col = n0 + t * 16 + l15;
                float bv = bias[col];
#pragma unroll
                for (int r = 0; r < 8; r++) {
                    int row = mb + r + 8 * hiw;
                    out[(size_t)row * 1024 + col] =
                        acc[mm][t][r] + bv + resid[(size_t)row * 1024 + col];
                }
            }
        } else {
            _Float16* out = (_Float16*)outp;
#pragma unroll
            for (int t = 0; t < 4; t++) {
                int col = n0 + t * 16 + l15;
                float bv = bias[col];
                int h = col >> 6, hd = col & 63;
#pragma unroll
                for (int r = 0; r < 8; r++) {
                    int row  = mb + r + 8 * hiw;
                    int bidx = row >> 9;
                    int pos  = row & 511;
                    float val = acc[mm][t][r] + bv;
                    if (MODE == 2) {
                        // V^T layout: [bh][hd][l]
                        out[(((size_t)bidx * NHEAD + h) * HDIM + hd) * SEQ + pos] =
                            (_Float16)val;
                    } else {
                        // RoPE: interleaved pairs (2i,2i+1); adjacent cols = adjacent lanes
                        float partner = __shfl_xor(val, 1, 32);
                        int i = hd >> 1;
                        // inv_freq[i] = 10000^(-2i/1024); 2*ln(10000)/1024 = 0.0179889460
                        float freq = __expf(-0.0179889460f * (float)i);
                        float ang = (float)pos * freq;
                        float sn, cs;
                        __sincosf(ang, &sn, &cs);
                        float o = val * cs + partner * (((col & 1) == 0) ? -sn : sn);
                        if (MODE == 0) o *= 0.125f;   // fold 1/sqrt(HD) into Q
                        out[(((size_t)bidx * NHEAD + h) * SEQ + pos) * HDIM + hd] =
                            (_Float16)o;
                    }
                }
            }
        }
    }
}

// ---------------------------------------------------------------------------
// Flash attention: one wave handles 16 query rows of one (b,h), all 512 keys,
// 32 keys per iteration. Q,K: f16 [bh][l][64]; Vt: f16 [bh][64][l].
// Output O: f16 [b*512+l][h*64+d] (GEMM-ready for the Wo projection).
// ---------------------------------------------------------------------------
__global__ __launch_bounds__(256) void attn_kernel(const _Float16* __restrict__ Q,
                                                   const _Float16* __restrict__ K,
                                                   const _Float16* __restrict__ Vt,
                                                   _Float16* __restrict__ O) {
    __shared__ _Float16 ldsP[8][16][32];   // per-wave 16x32 P tile (1KB each)
    const int lane = threadIdx.x & 31;
    const int wave = threadIdx.x >> 5;
    const int tile = blockIdx.x * 8 + wave;   // 0..4095
    const int bh = tile >> 5;                 // 0..127
    const int qt = tile & 31;                 // 0..31
    const int q0 = qt * 16;
    const int l15 = lane & 15;
    const int hiw = lane >> 4;
    const int kk = hiw * 16;

    const _Float16* Qh = Q  + (size_t)bh * SEQ * HDIM;
    const _Float16* Kh = K  + (size_t)bh * SEQ * HDIM;
    const _Float16* Vh = Vt + (size_t)bh * HDIM * SEQ;

    // Q A-fragments for d-chunks [0,32) and [32,64)
    AFrag qa0, qa1;
    {
        const _Float16* qp = Qh + (size_t)(q0 + l15) * HDIM + hiw * 16;
        qa0.h[0] = *(const v8h*)(qp);
        qa0.h[1] = *(const v8h*)(qp + 8);
        qa1.h[0] = *(const v8h*)(qp + 32);
        qa1.h[1] = *(const v8h*)(qp + 40);
    }

    float m_arr[8], l_arr[8];
    v8f acc[4];
#pragma unroll
    for (int r = 0; r < 8; r++) { m_arr[r] = -1e30f; l_arr[r] = 0.0f; }
#pragma unroll
    for (int t = 0; t < 4; t++) acc[t] = zerov8();

    for (int n0 = 0; n0 < SEQ; n0 += 32) {
        // K B-fragments: column = key row; per-lane 16 contiguous d values
        const _Float16* kp0 = Kh + (size_t)(n0 + l15) * HDIM + kk;
        const _Float16* kp1 = kp0 + 16 * HDIM;
        v16h kb00 = *(const v16h*)(kp0);
        v16h kb01 = *(const v16h*)(kp0 + 32);
        v16h kb10 = *(const v16h*)(kp1);
        v16h kb11 = *(const v16h*)(kp1 + 32);

        v8f s0 = zerov8(), s1 = zerov8();
        s0 = wmma_f16(qa0.v, kb00, s0);
        s0 = wmma_f16(qa1.v, kb01, s0);
        s1 = wmma_f16(qa0.v, kb10, s1);
        s1 = wmma_f16(qa1.v, kb11, s1);

        // Online softmax on 16x32 score tile (C layout: row spans 16 lanes)
#pragma unroll
        for (int r = 0; r < 8; r++) {
            float v0 = s0[r], v1 = s1[r];
            float mx = fmaxf(v0, v1);
#pragma unroll
            for (int off = 1; off < 16; off <<= 1)
                mx = fmaxf(mx, __shfl_xor(mx, off, 16));
            float mnew  = fmaxf(m_arr[r], mx);
            float alpha = __expf(m_arr[r] - mnew);
            float p0 = __expf(v0 - mnew);
            float p1 = __expf(v1 - mnew);
            float ps = p0 + p1;
#pragma unroll
            for (int off = 1; off < 16; off <<= 1)
                ps += __shfl_xor(ps, off, 16);
            l_arr[r] = l_arr[r] * alpha + ps;
            m_arr[r] = mnew;
#pragma unroll
            for (int t = 0; t < 4; t++) acc[t][r] *= alpha;
            int rowp = r + 8 * hiw;
            ldsP[wave][rowp][l15]      = (_Float16)p0;
            ldsP[wave][rowp][16 + l15] = (_Float16)p1;
        }

        // C-layout -> A-layout transpose through LDS (same-wave RAW)
        asm volatile("s_wait_dscnt 0" ::: "memory");
        AFrag pa;
        {
            const _Float16* pp = &ldsP[wave][l15][hiw * 16];
            pa.h[0] = *(const v8h*)(pp);
            pa.h[1] = *(const v8h*)(pp + 8);
        }

        // P(16x32) @ V(32x64): V^T layout makes B-fragments contiguous
#pragma unroll
        for (int t = 0; t < 4; t++) {
            const _Float16* vp = Vh + (size_t)(t * 16 + l15) * SEQ + n0 + kk;
            v16h vb = *(const v16h*)(vp);
            acc[t] = wmma_f16(pa.v, vb, acc[t]);
        }
    }

    const int b = bh >> 4, h = bh & 15;
#pragma unroll
    for (int t = 0; t < 4; t++) {
#pragma unroll
        for (int r = 0; r < 8; r++) {
            int row = b * SEQ + q0 + r + 8 * hiw;
            int col = h * HDIM + t * 16 + l15;
            O[(size_t)row * D_MODEL + col] = (_Float16)(acc[t][r] / l_arr[r]);
        }
    }
}

// ---------------------------------------------------------------------------
extern "C" void kernel_launch(void* const* d_in, const int* in_sizes, int n_in,
                              void* d_out, int out_size, void* d_ws, size_t ws_size,
                              hipStream_t stream) {
    (void)in_sizes; (void)n_in; (void)out_size; (void)ws_size;
    const float* input  = (const float*)d_in[0];
    const float* Wq     = (const float*)d_in[1];
    const float* bq     = (const float*)d_in[2];
    const float* Wk     = (const float*)d_in[3];
    const float* bk     = (const float*)d_in[4];
    const float* Wv     = (const float*)d_in[5];
    const float* bv     = (const float*)d_in[6];
    const float* Wo     = (const float*)d_in[7];
    const float* bo     = (const float*)d_in[8];
    const float* g_attn = (const float*)d_in[13];
    const float* b_attn = (const float*)d_in[14];
    float* out = (float*)d_out;

    char* ws = (char*)d_ws;
    const size_t WSZ = (size_t)1024 * 1024 * sizeof(_Float16);  // 2 MB unit
    _Float16* Wtq = (_Float16*)(ws + 0 * WSZ);
    _Float16* Wtk = (_Float16*)(ws + 1 * WSZ);
    _Float16* Wtv = (_Float16*)(ws + 2 * WSZ);
    _Float16* Wto = (_Float16*)(ws + 3 * WSZ);
    _Float16* xn  = (_Float16*)(ws + 4 * WSZ);   // 4096x1024 f16 = 8 MB
    _Float16* Qf  = (_Float16*)(ws + 8 * WSZ);   // [128][512][64]
    _Float16* Kf  = (_Float16*)(ws + 12 * WSZ);  // [128][512][64]
    _Float16* Vtf = (_Float16*)(ws + 16 * WSZ);  // [128][64][512]
    _Float16* Of  = (_Float16*)(ws + 20 * WSZ);  // [4096][1024]  (total 48 MB)

    wconv_kernel<<<4096, 256, 0, stream>>>(Wq, Wtq);
    wconv_kernel<<<4096, 256, 0, stream>>>(Wk, Wtk);
    wconv_kernel<<<4096, 256, 0, stream>>>(Wv, Wtv);
    wconv_kernel<<<4096, 256, 0, stream>>>(Wo, Wto);

    ln_kernel<<<MTOT, 256, 0, stream>>>(input, g_attn, b_attn, xn);

    gemm_kernel<0><<<256, 256, 0, stream>>>(xn, Wtq, bq, nullptr, (void*)Qf);
    gemm_kernel<1><<<256, 256, 0, stream>>>(xn, Wtk, bk, nullptr, (void*)Kf);
    gemm_kernel<2><<<256, 256, 0, stream>>>(xn, Wtv, bv, nullptr, (void*)Vtf);

    attn_kernel<<<512, 256, 0, stream>>>(Qf, Kf, Vtf, Of);

    gemm_kernel<3><<<256, 256, 0, stream>>>(Of, Wto, bo, input, (void*)out);
}
